// SioConvLayer_50775103373972
// MI455X (gfx1250) — matrix operand order repaired and checked by reading, not compiled, not run on hardware
//
#include <hip/hip_runtime.h>

// ---------------- problem constants (fixed by the reference) ----------------
#define B_     4
#define L_     4096
#define DIM_   1024
#define INNER_ 512
#define M_     (B_ * L_)   // 16384 rows in all GEMMs

// ---------------- WMMA types ----------------
typedef __attribute__((ext_vector_type(16))) __bf16 v16bf;
typedef __attribute__((ext_vector_type(8)))  float  v8f;
typedef __attribute__((ext_vector_type(4)))  int    v4i;

union FragU { uint4 q[2]; v16bf v; };

__device__ __forceinline__ unsigned short f2bf(float f) {
  union { float f; unsigned u; } x; x.f = f;
  unsigned r = x.u + 0x7FFFu + ((x.u >> 16) & 1u);   // round-to-nearest-even
  return (unsigned short)(r >> 16);
}

// ---------------- CDNA5 async global->LDS staging (ASYNCcnt path) ----------
#if defined(__has_builtin)
# if __has_builtin(__builtin_amdgcn_global_load_async_to_lds_b128)
#  define ASYNC_LDS 1
# endif
#endif

__device__ __forceinline__ void copy16_g2l(const unsigned short* g, unsigned short* l) {
#ifdef ASYNC_LDS
  __builtin_amdgcn_global_load_async_to_lds_b128(
      (__attribute__((address_space(1))) v4i*)g,
      (__attribute__((address_space(3))) v4i*)l, 0, 0);
#else
  *(uint4*)l = *(const uint4*)g;
#endif
}

__device__ __forceinline__ void async_wait_all() {
#ifdef ASYNC_LDS
# if __has_builtin(__builtin_amdgcn_s_wait_asynccnt)
  __builtin_amdgcn_s_wait_asynccnt(0);
# else
  asm volatile("s_wait_asynccnt 0" ::: "memory");
# endif
#endif
}

// =====================================================================
// f32 -> bf16 conversion (with optional sign, to pre-negate Vim/Vinv_im
// since bf16 WMMA has no A/B NEG modifier per the CDNA5 ISA)
// =====================================================================
__global__ void conv_f2bf(unsigned short* __restrict__ dst,
                          const float* __restrict__ src, int n, float scale) {
  int i = blockIdx.x * blockDim.x + threadIdx.x;
  if (i < n) dst[i] = f2bf(src[i] * scale);
}

// =====================================================================
// Gauss-Jordan inverse of complex 512x512 V (unnormalized elimination:
// row k is read-only during pivot k -> one race-free launch per pivot).
// Only columns [k, 512+k] are non-trivial at pivot k.
// =====================================================================
__global__ void gj_init(float* __restrict__ aug,
                        const float* __restrict__ Vre,
                        const float* __restrict__ Vim) {
  int idx = blockIdx.x * blockDim.x + threadIdx.x;   // over 512*1024 complex
  if (idx >= INNER_ * 2 * INNER_) return;
  int r = idx >> 10;
  int c = idx & 1023;
  float re, im;
  if (c < INNER_) { re = Vre[r * INNER_ + c]; im = Vim[r * INNER_ + c]; }
  else            { re = (c - INNER_ == r) ? 1.0f : 0.0f; im = 0.0f; }
  aug[2 * (size_t)idx]     = re;
  aug[2 * (size_t)idx + 1] = im;
}

__global__ void gj_elim(float* __restrict__ aug, int k) {
  const int W2 = 2048;                      // floats per row
  int r = blockIdx.x;
  if (r == k) return;
  float pr = aug[(size_t)k * W2 + 2 * k], pi = aug[(size_t)k * W2 + 2 * k + 1];
  float d  = pr * pr + pi * pi;
  float inr = pr / d, ini = -pi / d;        // 1/pivot
  float fr0 = aug[(size_t)r * W2 + 2 * k], fi0 = aug[(size_t)r * W2 + 2 * k + 1];
  float fr = fr0 * inr - fi0 * ini;
  float fi = fr0 * ini + fi0 * inr;
  for (int c = k + threadIdx.x; c <= INNER_ + k; c += blockDim.x) {
    float kr = aug[(size_t)k * W2 + 2 * c], ki = aug[(size_t)k * W2 + 2 * c + 1];
    float vr = aug[(size_t)r * W2 + 2 * c], vi = aug[(size_t)r * W2 + 2 * c + 1];
    aug[(size_t)r * W2 + 2 * c]     = vr - (fr * kr - fi * ki);
    aug[(size_t)r * W2 + 2 * c + 1] = vi - (fr * ki + fi * kr);
  }
}

__global__ void gj_finish(const float* __restrict__ aug,
                          float* __restrict__ vinv_re,
                          float* __restrict__ vinv_im) {
  const int W2 = 2048;
  int r = blockIdx.x;
  float pr = aug[(size_t)r * W2 + 2 * r], pi = aug[(size_t)r * W2 + 2 * r + 1];
  float d  = pr * pr + pi * pi;
  float inr = pr / d, ini = -pi / d;
  for (int c = threadIdx.x; c < INNER_; c += blockDim.x) {
    float re = aug[(size_t)r * W2 + 2 * (INNER_ + c)];
    float im = aug[(size_t)r * W2 + 2 * (INNER_ + c) + 1];
    vinv_re[r * INNER_ + c] = re * inr - im * ini;
    vinv_im[r * INNER_ + c] = re * ini + im * inr;
  }
}

// =====================================================================
// bf16-WMMA GEMM:  acc = A1 @ B1^T [+ A2 @ B2^T]   (signs pre-baked in B2)
// 128x128x32 tiles, 256 threads (8 waves), 8 v_wmma per wave per k-step.
// Double-buffered LDS fed by async global->LDS b128 copies; tile s+1 is
// prefetched while tile s is consumed (s_wait_asynccnt + barrier).
// Epilogue modes:
//   0: f32 store (+Eadd)                       -> p0
//   1: complex a-gate (rsqrt(m2)*m2/(1+m2))    -> p0=a_re f32, p1=a_im f32
//   2: complex x-gate (sqrt(m2))               -> p0=xc_re f32, p1/p2 bf16
//   3: +Eadd then bf16 store                   -> p0 bf16
// =====================================================================
#define BM 128
#define BN 128
#define BK 32

__global__ __launch_bounds__(256)
void gemm_bf(const unsigned short* __restrict__ A1, const unsigned short* __restrict__ B1,
             const unsigned short* __restrict__ A2, const unsigned short* __restrict__ B2,
             const unsigned short* __restrict__ h0a, const unsigned short* __restrict__ h0b,
             const float* __restrict__ Eadd,
             void* __restrict__ p0, void* __restrict__ p1, void* __restrict__ p2,
             int M, int N, int K, int shiftA, int mode) {
  __shared__ __align__(16) unsigned short As[2][BM][BK + 8];
  __shared__ __align__(16) unsigned short Bs[2][BN][BK + 8];

  const int tid  = threadIdx.x;
  const int wave = tid >> 5;
  const int lane = tid & 31;
  const int lr   = lane & 15;
  const int lh   = lane >> 4;
  const int wm   = wave >> 2;      // 0..1  -> 64-row slab
  const int wn   = wave & 3;       // 0..3  -> 32-col slab
  const int m0   = blockIdx.y * BM;
  const int n0   = blockIdx.x * BN;

  v8f zero = {0.f, 0.f, 0.f, 0.f, 0.f, 0.f, 0.f, 0.f};
  v8f acc[4][2];
  for (int i = 0; i < 4; ++i)
    for (int j = 0; j < 2; ++j) acc[i][j] = zero;

  const int stepsPerPh = K / BK;
  const int nph   = (A2 != nullptr) ? 2 : 1;
  const int total = stepsPerPh * nph;

  auto stage = [&](int sb, int s) {
    int ph = s / stepsPerPh;
    int k0 = (s - ph * stepsPerPh) * BK;
    const unsigned short* Ap = ph ? A2 : A1;
    const unsigned short* Bp = ph ? B2 : B1;
    const unsigned short* h0 = ph ? h0b : h0a;
    // 512 16-byte chunks per operand tile, 2 per thread
#pragma unroll
    for (int j = 0; j < 2; ++j) {
      int i = tid + 256 * j;
      int r = i >> 2, q = i & 3;
      int gm = m0 + r;
      unsigned short* la = &As[sb][r][q * 8];
      if (shiftA) {
        int t = gm & (L_ - 1);
        if (t == 0) *(uint4*)la = *(const uint4*)(h0 + k0 + q * 8);
        else        copy16_g2l(Ap + (size_t)(gm - 1) * K + k0 + q * 8, la);
      } else {
        copy16_g2l(Ap + (size_t)gm * K + k0 + q * 8, la);
      }
      copy16_g2l(Bp + (size_t)(n0 + r) * K + k0 + q * 8, &Bs[sb][r][q * 8]);
    }
  };

  stage(0, 0);
  for (int s = 0; s < total; ++s) {
    async_wait_all();
    __syncthreads();
    if (s + 1 < total) stage((s + 1) & 1, s + 1);

    const int sb = s & 1;
    FragU af[4], bfr[2];
#pragma unroll
    for (int mf = 0; mf < 4; ++mf) {
      // A (16x32): lane = row M, lo-lanes K {0..7,16..23}, hi-lanes +8
      const unsigned short* p = &As[sb][wm * 64 + mf * 16 + lr][lh * 8];
      af[mf].q[0] = *(const uint4*)p;
      af[mf].q[1] = *(const uint4*)(p + 16);
    }
#pragma unroll
    for (int nf = 0; nf < 2; ++nf) {
      // B (32x16): lane = col N, regs = K; lo-lanes K {0..7,16..23}, hi-lanes +8
      const unsigned short* p = &Bs[sb][wn * 32 + nf * 16 + lr][lh * 8];
      bfr[nf].q[0] = *(const uint4*)p;
      bfr[nf].q[1] = *(const uint4*)(p + 16);
    }
#pragma unroll
    for (int mf = 0; mf < 4; ++mf)
#pragma unroll
      for (int nf = 0; nf < 2; ++nf)
        acc[mf][nf] = __builtin_amdgcn_wmma_f32_16x16x32_bf16(
            false, af[mf].v, false, bfr[nf].v,
            (short)0, acc[mf][nf], false, false);
  }

  // ---- epilogue (C/D layout: VGPR r -> M = r (+8 hi lanes), lane -> N) ----
#pragma unroll
  for (int mf = 0; mf < 4; ++mf)
#pragma unroll
    for (int nf = 0; nf < 2; ++nf)
#pragma unroll
      for (int r = 0; r < 8; ++r) {
        int row = m0 + wm * 64 + mf * 16 + lh * 8 + r;
        int col = n0 + wn * 32 + nf * 16 + lr;
        float v = acc[mf][nf][r];
        if (mode == 0) {
          size_t idx = (size_t)row * N + col;
          if (Eadd) v += Eadd[idx];
          ((float*)p0)[idx] = v;
        } else if (mode == 3) {
          size_t idx = (size_t)row * N + col;
          v += Eadd[idx];
          ((unsigned short*)p0)[idx] = f2bf(v);
        } else {
          // complex channel: even lane = re, odd lane = im of channel col>>1
          float o  = __shfl_xor(v, 1, 32);
          float m2 = v * v + o * o;
          size_t idx = (size_t)row * INNER_ + (col >> 1);
          if (mode == 1) {                       // a-gate
            float sc = rsqrtf(m2) * (m2 / (1.0f + m2));
            float ov = v * sc;
            if ((lane & 1) == 0) ((float*)p0)[idx] = ov;   // a_re
            else                 ((float*)p1)[idx] = ov;   // a_im
          } else {                               // mode 2: x-gate
            float ov = v * sqrtf(m2);
            if ((lane & 1) == 0) {
              ((float*)p0)[idx] = ov;                       // xc_re (f32 Eadd)
              ((unsigned short*)p1)[idx] = f2bf(ov);        // xc_re bf16
            } else {
              ((unsigned short*)p2)[idx] = f2bf(ov);        // xc_im bf16
            }
          }
        }
      }
}

// =====================================================================
// Diagonal linear recurrence: s_t = a_t * (s_{t-1} + u_t)
// 2048 independent complex sequences; emits bf16 s for the V-basis GEMM.
// =====================================================================
__global__ void scan_kernel(const float* __restrict__ a_re, const float* __restrict__ a_im,
                            const float* __restrict__ u_re, const float* __restrict__ u_im,
                            unsigned short* __restrict__ s_re, unsigned short* __restrict__ s_im) {
  int g = blockIdx.x * blockDim.x + threadIdx.x;   // 0..2047
  int b = g >> 9;
  int c = g & (INNER_ - 1);
  size_t base = (size_t)b * L_ * INNER_ + c;
  float sr = 0.f, si = 0.f;
  for (int t = 0; t < L_; ++t) {
    size_t o = base + (size_t)t * INNER_;
    float tr = sr + u_re[o];
    float ti = si + u_im[o];
    float ar = a_re[o], ai = a_im[o];
    sr = ar * tr - ai * ti;
    si = ar * ti + ai * tr;
    s_re[o] = f2bf(sr);
    s_im[o] = f2bf(si);
  }
}

// =====================================================================
// Host-side orchestration (stream launches only; graph-capture safe)
// =====================================================================
extern "C" void kernel_launch(void* const* d_in, const int* in_sizes, int n_in,
                              void* d_out, int out_size, void* d_ws, size_t ws_size,
                              hipStream_t stream) {
  (void)in_sizes; (void)n_in; (void)out_size; (void)ws_size;
  const float* x     = (const float*)d_in[0];
  const float* w_a   = (const float*)d_in[1];
  const float* w_x   = (const float*)d_in[2];
  const float* w_out = (const float*)d_in[3];
  const float* Vre   = (const float*)d_in[4];
  const float* Vim   = (const float*)d_in[5];
  const float* h0re  = (const float*)d_in[6];
  const float* h0im  = (const float*)d_in[7];
  float* out = (float*)d_out;

  // ---- workspace layout (float units; ushort buffers counted as n/2) ----
  float* ws = (float*)d_ws;
  size_t o = 0;
  float* aug      = ws + o; o += (size_t)2 * INNER_ * 1024;       // 4 MB
  float* vinv_ref = ws + o; o += (size_t)INNER_ * INNER_;
  float* vinv_imf = ws + o; o += (size_t)INNER_ * INNER_;
  unsigned short* x_bf       = (unsigned short*)(ws + o); o += (size_t)M_ * DIM_ / 2;
  unsigned short* wa_bf      = (unsigned short*)(ws + o); o += (size_t)DIM_ * DIM_ / 2;
  unsigned short* wx_bf      = (unsigned short*)(ws + o); o += (size_t)DIM_ * DIM_ / 2;
  unsigned short* wout_bf    = (unsigned short*)(ws + o); o += (size_t)DIM_ * INNER_ / 2;
  unsigned short* vre_bf     = (unsigned short*)(ws + o); o += (size_t)INNER_ * INNER_ / 2;
  unsigned short* vimn_bf    = (unsigned short*)(ws + o); o += (size_t)INNER_ * INNER_ / 2;
  unsigned short* vinvre_bf  = (unsigned short*)(ws + o); o += (size_t)INNER_ * INNER_ / 2;
  unsigned short* vinvim_bf  = (unsigned short*)(ws + o); o += (size_t)INNER_ * INNER_ / 2;
  unsigned short* vinvimn_bf = (unsigned short*)(ws + o); o += (size_t)INNER_ * INNER_ / 2;
  unsigned short* h0re_bf    = (unsigned short*)(ws + o); o += 256;
  unsigned short* h0im_bf    = (unsigned short*)(ws + o); o += 256;
  float* a_re  = ws + o; o += (size_t)M_ * INNER_;
  float* a_im  = ws + o; o += (size_t)M_ * INNER_;
  float* xc_re = ws + o; o += (size_t)M_ * INNER_;                // f32 (Eadd)
  unsigned short* xcre_bf = (unsigned short*)(ws + o); o += (size_t)M_ * INNER_ / 2;
  unsigned short* xcim_bf = (unsigned short*)(ws + o); o += (size_t)M_ * INNER_ / 2;
  float* u_re  = ws + o; o += (size_t)M_ * INNER_;
  float* u_im  = ws + o; o += (size_t)M_ * INNER_;
  unsigned short* sre_bf = (unsigned short*)(ws + o); o += (size_t)M_ * INNER_ / 2;
  unsigned short* sim_bf = (unsigned short*)(ws + o); o += (size_t)M_ * INNER_ / 2;
  unsigned short* h_bf   = (unsigned short*)u_re;   // reuse u region post-scan

  dim3 blk(256);
  const int NN = INNER_ * INNER_;

  // 1) Vinv = inv(V) (one race-free elimination launch per pivot)
  gj_init<<<dim3((INNER_ * 2 * INNER_) / 256), blk, 0, stream>>>(aug, Vre, Vim);
  for (int k = 0; k < INNER_; ++k)
    gj_elim<<<dim3(INNER_), blk, 0, stream>>>(aug, k);
  gj_finish<<<dim3(INNER_), blk, 0, stream>>>(aug, vinv_ref, vinv_imf);

  // 2) pre-convert all GEMM operands to bf16 (signs baked into *_neg copies)
  conv_f2bf<<<dim3((M_ * DIM_) / 256), blk, 0, stream>>>(x_bf, x, M_ * DIM_, 1.f);
  conv_f2bf<<<dim3((DIM_ * DIM_) / 256), blk, 0, stream>>>(wa_bf, w_a, DIM_ * DIM_, 1.f);
  conv_f2bf<<<dim3((DIM_ * DIM_) / 256), blk, 0, stream>>>(wx_bf, w_x, DIM_ * DIM_, 1.f);
  conv_f2bf<<<dim3((DIM_ * INNER_) / 256), blk, 0, stream>>>(wout_bf, w_out, DIM_ * INNER_, 1.f);
  conv_f2bf<<<dim3(NN / 256), blk, 0, stream>>>(vre_bf, Vre, NN, 1.f);
  conv_f2bf<<<dim3(NN / 256), blk, 0, stream>>>(vimn_bf, Vim, NN, -1.f);
  conv_f2bf<<<dim3(NN / 256), blk, 0, stream>>>(vinvre_bf, vinv_ref, NN, 1.f);
  conv_f2bf<<<dim3(NN / 256), blk, 0, stream>>>(vinvim_bf, vinv_imf, NN, 1.f);
  conv_f2bf<<<dim3(NN / 256), blk, 0, stream>>>(vinvimn_bf, vinv_imf, NN, -1.f);
  conv_f2bf<<<dim3(2), blk, 0, stream>>>(h0re_bf, h0re, INNER_, 1.f);
  conv_f2bf<<<dim3(2), blk, 0, stream>>>(h0im_bf, h0im, INNER_, 1.f);

  // 3) projections with fused complex gates in the epilogue
  gemm_bf<<<dim3(DIM_ / BN, M_ / BM), blk, 0, stream>>>(        // a-gate
      x_bf, wa_bf, nullptr, nullptr, nullptr, nullptr, nullptr,
      a_re, a_im, nullptr, M_, DIM_, DIM_, 0, 1);
  gemm_bf<<<dim3(DIM_ / BN, M_ / BM), blk, 0, stream>>>(        // x-gate
      x_bf, wx_bf, nullptr, nullptr, nullptr, nullptr, nullptr,
      xc_re, xcre_bf, xcim_bf, M_, DIM_, DIM_, 0, 2);

  // 4) u = Vinv @ x_roll (complex; shift + h0 injection in the A loader)
  gemm_bf<<<dim3(INNER_ / BN, M_ / BM), blk, 0, stream>>>(      // u_re
      xcre_bf, vinvre_bf, xcim_bf, vinvimn_bf, h0re_bf, h0im_bf, nullptr,
      u_re, nullptr, nullptr, M_, INNER_, INNER_, 1, 0);
  gemm_bf<<<dim3(INNER_ / BN, M_ / BM), blk, 0, stream>>>(      // u_im
      xcre_bf, vinvim_bf, xcim_bf, vinvre_bf, h0re_bf, h0im_bf, nullptr,
      u_im, nullptr, nullptr, M_, INNER_, INNER_, 1, 0);

  // 5) recurrence scan (emits bf16 s)
  scan_kernel<<<dim3((B_ * INNER_) / 256), blk, 0, stream>>>(
      a_re, a_im, u_re, u_im, sre_bf, sim_bf);

  // 6) h_re = s_re@Vre^T - s_im@Vim^T + xc_re  -> bf16 h
  gemm_bf<<<dim3(INNER_ / BN, M_ / BM), blk, 0, stream>>>(
      sre_bf, vre_bf, sim_bf, vimn_bf, nullptr, nullptr, xc_re,
      h_bf, nullptr, nullptr, M_, INNER_, INNER_, 0, 3);

  // 7) out = h @ w_out^T
  gemm_bf<<<dim3(DIM_ / BN, M_ / BM), blk, 0, stream>>>(
      h_bf, wout_bf, nullptr, nullptr, nullptr, nullptr, nullptr,
      out, nullptr, nullptr, M_, DIM_, INNER_, 0, 0);
}
